// Attention_xxc_37409165148275
// MI455X (gfx1250) — matrix operand order repaired and checked by reading, not compiled, not run on hardware
//
#include <hip/hip_runtime.h>

typedef unsigned short ushort_t;
typedef unsigned int uint_t;
typedef __attribute__((ext_vector_type(16))) __bf16 v16bf;
typedef __attribute__((ext_vector_type(8))) float v8f;

#define DIM       512
#define NUM_HEADS 8
#define HEAD_DIM  64
#define BATCH     16
#define NTOK      1024
#define KHOPS     5
#define ATT_SCALE 0.125f  // 64^-0.5

union BF16Frag { v16bf v; uint_t u[8]; };
union F8       { v8f v; float f[8]; };

__device__ __forceinline__ ushort_t f2bf(float f) {
    uint_t u = __float_as_uint(f);
    u += 0x7FFFu + ((u >> 16) & 1u);      // round-to-nearest-even
    return (ushort_t)(u >> 16);
}

// A-operand pair mapping used here (contiguous k, B loads use the same): pair p of lane l holds K = k0, k0+1 (k0 = 2p + 16*(l/16))
__device__ __forceinline__ int a_k0(int p, int lane) {
    return 2 * p + ((lane >= 16) ? 16 : 0);
}

// LDS byte offset of a __shared__ pointer: low 32 bits of the generic address
__device__ __forceinline__ uint_t lds_off(const void* p) {
    return (uint_t)(size_t)p;
}

// Async global->LDS 16-byte copy (GVS mode: SGPR64 base + VGPR32 byte offset).
// Tracked by ASYNCcnt; data visible in LDS once the counter drains.
__device__ __forceinline__ void async_copy_b128(uint_t lds_byte, uint_t glob_byte,
                                                const void* base) {
    asm volatile("global_load_async_to_lds_b128 %0, %1, %2"
                 :: "v"(lds_byte), "v"(glob_byte),
                    "s"((unsigned long long)(size_t)base)
                 : "memory");
}

template <int N>
__device__ __forceinline__ void wait_async() {
    asm volatile("s_wait_asynccnt %0" :: "i"(N) : "memory");
}

// ---------------------------------------------------------------------------
// fp32 -> bf16 elementwise
// ---------------------------------------------------------------------------
__global__ void f32_to_bf16_kernel(const float* __restrict__ src,
                                   ushort_t* __restrict__ dst, int n) {
    int i = blockIdx.x * blockDim.x + threadIdx.x;
    if (i < n) dst[i] = f2bf(src[i]);
}

// ---------------------------------------------------------------------------
// bias[h,i,j] = rel_alpha[h] * sum_k softmax(hop_logits[h,:])[k] * Hstack[k,i,j]
// ---------------------------------------------------------------------------
__global__ void bias_kernel(const float* __restrict__ Hstack,
                            const float* __restrict__ hop_logits,
                            const float* __restrict__ rel_alpha,
                            float* __restrict__ bias) {
    const int h = blockIdx.y;
    const size_t NN = (size_t)NTOK * NTOK;
    float lg[KHOPS];
    float mx = -3.0e38f;
#pragma unroll
    for (int k = 0; k < KHOPS; ++k) {
        lg[k] = hop_logits[h * KHOPS + k];
        mx = fmaxf(mx, lg[k]);
    }
    float s = 0.0f;
#pragma unroll
    for (int k = 0; k < KHOPS; ++k) { lg[k] = __expf(lg[k] - mx); s += lg[k]; }
    const float inv = 1.0f / s;
    const float alpha = rel_alpha[h];
    const size_t idx = (size_t)blockIdx.x * blockDim.x + threadIdx.x;
    float acc = 0.0f;
#pragma unroll
    for (int k = 0; k < KHOPS; ++k) acc += lg[k] * inv * Hstack[k * NN + idx];
    bias[h * NN + idx] = alpha * acc;
}

// ---------------------------------------------------------------------------
// Shared GEMM core: C[128x128] = A[128x512] * B[128x512]^T (both bf16,
// row-major, K contiguous). 256 threads = 8 waves; wave w owns rows
// [w*16,w*16+16) x 128 cols. Double-buffered async global->LDS staging.
// As/Bs each hold two 128x32 bf16 tiles (2 x 8KB).
// ---------------------------------------------------------------------------
__device__ __forceinline__ void gemm_core_128x128(
    const ushort_t* __restrict__ Ag, const ushort_t* __restrict__ Bg,
    int m0, int n0, ushort_t* As, ushort_t* Bs, F8 acc[8]) {
    const int tid = threadIdx.x;
    const int wave = tid >> 5;
    const int lane = tid & 31;
    const int lmod = lane & 15;
    const int lhi = (lane >> 4) & 1;
    const uint_t* Asu = (const uint_t*)As;  // 16 uints per row (32 bf16)
    const uint_t* Bsu = (const uint_t*)Bs;
    const uint_t asBase = lds_off(As);
    const uint_t bsBase = lds_off(Bs);

    // Per-thread staging slot: row r = tid/2, uints [cu, cu+8) of the 32-bf16
    // row (two b128 transfers). LDS slot = tid*32 bytes within the tile.
    const int r = tid >> 1;
    const int cu = (tid & 1) * 8;

#pragma unroll
    for (int t = 0; t < 8; ++t)
#pragma unroll
        for (int j = 0; j < 8; ++j) acc[t].f[j] = 0.0f;

    // issue tile i into buffer (i&1)
    auto issue_tile = [&](int kk, int sel) {
        const uint_t ku0 = (uint_t)(kk >> 1);
        const uint_t ldsA = asBase + (uint_t)sel * 8192u + (uint_t)tid * 32u;
        const uint_t ldsB = bsBase + (uint_t)sel * 8192u + (uint_t)tid * 32u;
        const uint_t ga = ((uint_t)(m0 + r) * 256u + ku0 + (uint_t)cu) * 4u;
        const uint_t gb = ((uint_t)(n0 + r) * 256u + ku0 + (uint_t)cu) * 4u;
        async_copy_b128(ldsA, ga, Ag);
        async_copy_b128(ldsA + 16u, ga + 16u, Ag);
        async_copy_b128(ldsB, gb, Bg);
        async_copy_b128(ldsB + 16u, gb + 16u, Bg);
    };

    issue_tile(0, 0);
    const int NT = DIM / 32;  // 16 k-steps
    for (int i = 0; i < NT; ++i) {
        __syncthreads();  // all waves done computing tile i-1 (buffer (i+1)&1)
        if (i + 1 < NT) {
            issue_tile((i + 1) * 32, (i + 1) & 1);
            wait_async<4>();  // in-order completion: tile i (first 4 ops) landed
        } else {
            wait_async<0>();
        }
        __syncthreads();  // tile i visible to every wave

        const int sel = (i & 1) * 2048;  // uint offset of current buffer
        BF16Frag af;
        const int arow = wave * 16 + lmod;
#pragma unroll
        for (int p = 0; p < 8; ++p)
            af.u[p] = Asu[sel + arow * 16 + (a_k0(p, lane) >> 1)];

#pragma unroll
        for (int t = 0; t < 8; ++t) {
            BF16Frag bfr;
            const int bofs = sel + (t * 16 + lmod) * 16 + (lhi ? 8 : 0);
#pragma unroll
            for (int p = 0; p < 8; ++p) bfr.u[p] = Bsu[bofs + p];
            acc[t].v = __builtin_amdgcn_wmma_f32_16x16x32_bf16(
                false, af.v, false, bfr.v, (short)0, acc[t].v, false, false);
        }
    }
}

// ---------------------------------------------------------------------------
// QKV projection: qkv = x @ Wqkv^T, scattered as bf16 Q/K/V [B,H,N,64]
// ---------------------------------------------------------------------------
__global__ __launch_bounds__(256) void qkv_gemm_kernel(
    const ushort_t* __restrict__ xbf, const ushort_t* __restrict__ wbf,
    ushort_t* __restrict__ Q, ushort_t* __restrict__ K, ushort_t* __restrict__ V) {
    __shared__ __align__(16) ushort_t As[2 * 128 * 32];
    __shared__ __align__(16) ushort_t Bs[2 * 128 * 32];
    F8 acc[8];
    const int m0 = blockIdx.y * 128, n0 = blockIdx.x * 128;
    gemm_core_128x128(xbf, wbf, m0, n0, As, Bs, acc);

    const int tid = threadIdx.x, wave = tid >> 5, lane = tid & 31;
    const int lmod = lane & 15, lhi = (lane >> 4) & 1;
#pragma unroll
    for (int t = 0; t < 8; ++t)
#pragma unroll
        for (int v = 0; v < 8; ++v) {
            int m = m0 + wave * 16 + v + lhi * 8;
            int n = n0 + t * 16 + lmod;
            int which = n >> 9;
            int h = (n >> 6) & 7;
            int d = n & 63;
            int b = m >> 10;
            int tok = m & 1023;
            size_t dst = ((size_t)(b * NUM_HEADS + h) * NTOK + tok) * HEAD_DIM + d;
            ushort_t val = f2bf(acc[t].f[v]);
            if (which == 0) Q[dst] = val;
            else if (which == 1) K[dst] = val;
            else V[dst] = val;
        }
}

// ---------------------------------------------------------------------------
// Flash attention: grid (N/64, H, B), 128 threads = 4 waves, each wave a
// 16-query tile. Streams K/V in 32-key blocks through LDS (K via async
// global->LDS DMA), online softmax.
// Output written bf16 as [B*N, H*64+d] (= transpose(0,2,1,3).reshape).
// ---------------------------------------------------------------------------
__global__ __launch_bounds__(128) void attention_kernel(
    const ushort_t* __restrict__ Q, const ushort_t* __restrict__ K,
    const ushort_t* __restrict__ V, const float* __restrict__ bias,
    ushort_t* __restrict__ O) {
    __shared__ __align__(16) ushort_t KL[32 * 64];      // [key][d]
    __shared__ __align__(16) ushort_t VT[64 * 32];      // [d][key]
    __shared__ __align__(16) ushort_t PL[4 * 16 * 32];  // per-wave [row][key]

    const int qb = blockIdx.x, h = blockIdx.y, b = blockIdx.z;
    const int tid = threadIdx.x, wave = tid >> 5, lane = tid & 31;
    const int lmod = lane & 15, lhi = (lane >> 4) & 1;
    const size_t bh = (size_t)(b * NUM_HEADS + h);

    const uint_t* Qu = (const uint_t*)Q;
    const uint_t* KLu = (const uint_t*)KL;
    const uint_t* VTu = (const uint_t*)VT;
    const uint_t* PLu = (const uint_t*)PL + wave * 256;  // 16 rows * 16 uints
    const uint_t klBase = lds_off(KL);

    // Q fragments (16 rows x 64 K, two 16x32 chunks), 32 uints per row
    BF16Frag aq[2];
    {
        const size_t qrow = (bh * NTOK + (size_t)qb * 64 + wave * 16 + lmod) * 32;
#pragma unroll
        for (int c = 0; c < 2; ++c)
#pragma unroll
            for (int p = 0; p < 8; ++p)
                aq[c].u[p] = Qu[qrow + c * 16 + (a_k0(p, lane) >> 1)];
    }

    F8 o[4];
    float m_run[8], l_run[8];
#pragma unroll
    for (int v = 0; v < 8; ++v) { m_run[v] = -3.0e38f; l_run[v] = 0.0f; }
#pragma unroll
    for (int t = 0; t < 4; ++t)
#pragma unroll
        for (int v = 0; v < 8; ++v) o[t].f[v] = 0.0f;

    const float* biasH = bias + (size_t)h * NTOK * NTOK;

    for (int kb = 0; kb < NTOK / 32; ++kb) {
        __syncthreads();
        const uint_t kbyte = (uint_t)((bh * NTOK + (size_t)kb * 32) * 128);
        // K block: verbatim 4KB copy via async DMA (2 x b128 per thread)
        {
            const uint_t lk = klBase + (uint_t)tid * 32u;
            const uint_t gk = kbyte + (uint_t)tid * 32u;
            async_copy_b128(lk, gk, K);
            async_copy_b128(lk + 16u, gk + 16u, K);
        }
        // V block transposed into VT[d][key] (b128 loads, b16 scatter stores)
        {
            const uint4* Vg4 = (const uint4*)(V + (size_t)kbyte / 2);
#pragma unroll
            for (int j = 0; j < 2; ++j) {
                int i4 = tid + j * 128;        // 0..255 uint4 slots
                int key = i4 >> 3;
                int c4 = i4 & 7;               // covers d = c4*8 .. c4*8+7
                uint4 val = Vg4[i4];
                uint_t w[4] = {val.x, val.y, val.z, val.w};
#pragma unroll
                for (int q = 0; q < 4; ++q) {
                    int d0 = c4 * 8 + q * 2;
                    VT[d0 * 32 + key] = (ushort_t)(w[q] & 0xFFFFu);
                    VT[(d0 + 1) * 32 + key] = (ushort_t)(w[q] >> 16);
                }
            }
        }
        wait_async<0>();
        __syncthreads();

        // ---- S = Q @ K^T  (two 16x16 key tiles) ----
        F8 s[2];
#pragma unroll
        for (int t = 0; t < 2; ++t) {
            F8 z;
#pragma unroll
            for (int j = 0; j < 8; ++j) z.f[j] = 0.0f;
            BF16Frag b0, b1;
            // KL uint view: row = key (32 uints). Chunk c covers d in
            // [32c,32c+32): uint idx = key*32 + c*16 + p + (lhi?8:0)
            const int kbase2 = (t * 16 + lmod) * 32 + (lhi ? 8 : 0);
#pragma unroll
            for (int p = 0; p < 8; ++p) {
                b0.u[p] = KLu[kbase2 + p];
                b1.u[p] = KLu[kbase2 + 16 + p];
            }
            v8f cc = __builtin_amdgcn_wmma_f32_16x16x32_bf16(
                false, aq[0].v, false, b0.v, (short)0, z.v, false, false);
            cc = __builtin_amdgcn_wmma_f32_16x16x32_bf16(
                false, aq[1].v, false, b1.v, (short)0, cc, false, false);
            s[t].v = cc;
        }

        // ---- scale + bias, online softmax ----
        float rmax[8];
#pragma unroll
        for (int v = 0; v < 8; ++v) {
            size_t qr = (size_t)(qb * 64 + wave * 16 + v + lhi * 8) * NTOK +
                        (size_t)kb * 32 + lmod;
            float s0 = s[0].f[v] * ATT_SCALE + biasH[qr];
            float s1 = s[1].f[v] * ATT_SCALE + biasH[qr + 16];
            s[0].f[v] = s0;
            s[1].f[v] = s1;
            rmax[v] = fmaxf(s0, s1);
        }
#pragma unroll
        for (int off = 1; off < 16; off <<= 1)
#pragma unroll
            for (int v = 0; v < 8; ++v)
                rmax[v] = fmaxf(rmax[v], __shfl_xor(rmax[v], off, 32));

        float rsum[8];
#pragma unroll
        for (int v = 0; v < 8; ++v) {
            float mn = fmaxf(m_run[v], rmax[v]);
            float corr = __expf(m_run[v] - mn);
            float p0 = __expf(s[0].f[v] - mn);
            float p1 = __expf(s[1].f[v] - mn);
            const int prow = v + lhi * 8;
            PL[wave * 512 + prow * 32 + lmod] = f2bf(p0);
            PL[wave * 512 + prow * 32 + 16 + lmod] = f2bf(p1);
            rsum[v] = p0 + p1;
            m_run[v] = mn;
            l_run[v] *= corr;
#pragma unroll
            for (int t = 0; t < 4; ++t) o[t].f[v] *= corr;
        }
#pragma unroll
        for (int off = 1; off < 16; off <<= 1)
#pragma unroll
            for (int v = 0; v < 8; ++v) rsum[v] += __shfl_xor(rsum[v], off, 32);
#pragma unroll
        for (int v = 0; v < 8; ++v) l_run[v] += rsum[v];

        // wave-local LDS write->read of P: be explicit about the DS counter
        asm volatile("s_wait_dscnt 0" ::: "memory");

        // ---- O += P @ V  (A = P 16x32 keys, B = V 32x16 per d-tile) ----
        BF16Frag pf;
#pragma unroll
        for (int p = 0; p < 8; ++p)
            pf.u[p] = PLu[lmod * 16 + (a_k0(p, lane) >> 1)];
#pragma unroll
        for (int t = 0; t < 4; ++t) {
            BF16Frag vb;
            const int vbase = (t * 16 + lmod) * 16 + (lhi ? 8 : 0);
#pragma unroll
            for (int p = 0; p < 8; ++p) vb.u[p] = VTu[vbase + p];
            o[t].v = __builtin_amdgcn_wmma_f32_16x16x32_bf16(
                false, pf.v, false, vb.v, (short)0, o[t].v, false, false);
        }
    }

    // ---- epilogue: O /= l, write bf16 [B*N, DIM] with col = h*64+d ----
#pragma unroll
    for (int v = 0; v < 8; ++v) {
        float inv = 1.0f / l_run[v];
        size_t row = (size_t)b * NTOK + (size_t)qb * 64 + wave * 16 + v + lhi * 8;
#pragma unroll
        for (int t = 0; t < 4; ++t)
            O[row * DIM + h * 64 + t * 16 + lmod] = f2bf(o[t].f[v] * inv);
    }
}

// ---------------------------------------------------------------------------
// Output projection: out = attn_out @ Wproj^T + bproj  (fp32 out)
// ---------------------------------------------------------------------------
__global__ __launch_bounds__(256) void proj_gemm_kernel(
    const ushort_t* __restrict__ Abf, const ushort_t* __restrict__ Wbf,
    const float* __restrict__ bproj, float* __restrict__ out) {
    __shared__ __align__(16) ushort_t As[2 * 128 * 32];
    __shared__ __align__(16) ushort_t Bs[2 * 128 * 32];
    F8 acc[8];
    const int m0 = blockIdx.y * 128, n0 = blockIdx.x * 128;
    gemm_core_128x128(Abf, Wbf, m0, n0, As, Bs, acc);

    const int tid = threadIdx.x, wave = tid >> 5, lane = tid & 31;
    const int lmod = lane & 15, lhi = (lane >> 4) & 1;
#pragma unroll
    for (int t = 0; t < 8; ++t)
#pragma unroll
        for (int v = 0; v < 8; ++v) {
            int m = m0 + wave * 16 + v + lhi * 8;
            int n = n0 + t * 16 + lmod;
            out[(size_t)m * DIM + n] = acc[t].f[v] + bproj[n];
        }
}

// ---------------------------------------------------------------------------
extern "C" void kernel_launch(void* const* d_in, const int* in_sizes, int n_in,
                              void* d_out, int out_size, void* d_ws, size_t ws_size,
                              hipStream_t stream) {
    const float* x = (const float*)d_in[0];
    const float* Hstack = (const float*)d_in[1];
    const float* hop_logits = (const float*)d_in[2];
    const float* rel_alpha = (const float*)d_in[3];
    const float* Wqkv = (const float*)d_in[4];
    const float* Wproj = (const float*)d_in[5];
    const float* bproj = (const float*)d_in[6];
    float* out = (float*)d_out;

    char* ws = (char*)d_ws;
    size_t off = 0;
    float* bias = (float*)(ws + off);
    off += (size_t)NUM_HEADS * NTOK * NTOK * sizeof(float);        // 32 MB
    ushort_t* xbf = (ushort_t*)(ws + off);
    off += (size_t)BATCH * NTOK * DIM * 2;                         // 16 MB
    ushort_t* wqkvbf = (ushort_t*)(ws + off);
    off += (size_t)3 * DIM * DIM * 2;                              // 1.5 MB
    ushort_t* wprojbf = (ushort_t*)(ws + off);
    off += (size_t)DIM * DIM * 2;                                  // 0.5 MB
    const size_t qkvBytes = (size_t)BATCH * NUM_HEADS * NTOK * HEAD_DIM * 2;
    ushort_t* Qb = (ushort_t*)(ws + off); off += qkvBytes;         // 16 MB
    ushort_t* Kb = (ushort_t*)(ws + off); off += qkvBytes;         // 16 MB
    ushort_t* Vb = (ushort_t*)(ws + off); off += qkvBytes;         // 16 MB
    ushort_t* attout = (ushort_t*)(ws + off); off += qkvBytes;     // 16 MB

    // 1) conversions to bf16
    const int nx = BATCH * NTOK * DIM;  // 8388608
    f32_to_bf16_kernel<<<nx / 256, 256, 0, stream>>>(x, xbf, nx);
    const int nq = 3 * DIM * DIM;       // 786432
    f32_to_bf16_kernel<<<nq / 256, 256, 0, stream>>>(Wqkv, wqkvbf, nq);
    const int np = DIM * DIM;           // 262144
    f32_to_bf16_kernel<<<np / 256, 256, 0, stream>>>(Wproj, wprojbf, np);

    // 2) hop-bias precompute (32 MB fp32, stays hot in the 192 MB L2)
    bias_kernel<<<dim3(NTOK * NTOK / 256, NUM_HEADS), 256, 0, stream>>>(
        Hstack, hop_logits, rel_alpha, bias);

    // 3) QKV projection GEMM
    qkv_gemm_kernel<<<dim3(3 * DIM / 128, BATCH * NTOK / 128), 256, 0, stream>>>(
        xbf, wqkvbf, Qb, Kb, Vb);

    // 4) flash attention
    attention_kernel<<<dim3(NTOK / 64, NUM_HEADS, BATCH), 128, 0, stream>>>(
        Qb, Kb, Vb, bias, attout);

    // 5) output projection + bias
    proj_gemm_kernel<<<dim3(DIM / 128, BATCH * NTOK / 128), 256, 0, stream>>>(
        attout, wprojbf, bproj, out);
}
